// WindowAttentionConvRpe_65146063946591
// MI455X (gfx1250) — matrix operand order repaired and checked by reading, not compiled, not run on hardware
//
#include <hip/hip_runtime.h>

typedef _Float16 h16;
typedef __attribute__((ext_vector_type(16))) _Float16 v16h;
typedef __attribute__((ext_vector_type(8)))  _Float16 v8h;
typedef __attribute__((ext_vector_type(4)))  _Float16 v4h;
typedef __attribute__((ext_vector_type(8)))  float    v8f;
typedef __attribute__((ext_vector_type(4)))  float    v4f;

#define N_TOK 49
#define T_RPE 169
#define NHEAD 6
#define HDIM  32
#define CDIM  192
#define NWIN  64
#define BWIN  4096
#define NROW  (BWIN * N_TOK)          /* 200704 token rows */
#define VSTR  56                      /* padded v^T row stride in halves (112B, 16B-aligned) */
#define QK_SCALE 0.17677669529663687f /* 32^-0.5 */

// ---------------------------------------------------------------------------
// CDNA5 async global->LDS copy (ASYNCcnt path) + wait, via inline asm
// ---------------------------------------------------------------------------
__device__ __forceinline__ void async_b128(unsigned lds_dst, const void* src) {
  asm volatile("global_load_async_to_lds_b128 %0, %1, off"
               :: "v"(lds_dst), "v"((unsigned long long)(size_t)src)
               : "memory");
}
__device__ __forceinline__ void wait_async0() {
  asm volatile("s_wait_asynccnt 0x0" ::: "memory");
}

// ---------------------------------------------------------------------------
// WMMA fragment helpers (CDNA5 16x16x32 f16 layouts, wave32)
// A (16x32): lane r=lane&15 holds row M=r; lo lanes K{0..7,16..23}, hi lanes K{8..15,24..31}
// B (32x16): B^T (NxK) row-major in LDS; lane c=lane&15 holds col N=c; lo lanes K{0..7,16..23}, hi lanes K{8..15,24..31}
// C/D (16x16 f32): lane = 16*(m>=8)+n, vgpr = m&7
// ---------------------------------------------------------------------------
__device__ __forceinline__ v16h frag_a(const h16* lds, int stride, int row0, int k0, int lane) {
  int r  = lane & 15;
  int kb = (lane >> 4) << 3;
  const h16* p = lds + (row0 + r) * stride + k0 + kb;
  v8h lo = *(const v8h*)p;
  v8h hi = *(const v8h*)(p + 16);
  return __builtin_shufflevector(lo, hi, 0, 1, 2, 3, 4, 5, 6, 7,
                                 8, 9, 10, 11, 12, 13, 14, 15);
}

__device__ __forceinline__ v16h frag_b(const h16* lds, int stride, int col0, int k0, int lane) {
  int c  = lane & 15;
  int kb = (lane >> 4) << 3;
  const h16* p = lds + (col0 + c) * stride + k0 + kb;
  v8h lo = *(const v8h*)p;
  v8h hi = *(const v8h*)(p + 16);
  return __builtin_shufflevector(lo, hi, 0, 1, 2, 3, 4, 5, 6, 7,
                                 8, 9, 10, 11, 12, 13, 14, 15);
}

__device__ __forceinline__ v8f wmma_f16(v16h a, v16h b, v8f c) {
  return __builtin_amdgcn_wmma_f32_16x16x32_f16(false, a, false, b,
                                                (short)0, c, false, false);
}

__device__ __forceinline__ void store_d_h(h16* lds, int stride, int row0, int col0,
                                          v8f d, int lane, int rmax) {
  int c  = lane & 15;
  int mb = (lane >> 4) << 3;
  #pragma unroll
  for (int v = 0; v < 8; ++v) {
    int r = row0 + mb + v;
    if (r < rmax) lds[r * stride + col0 + c] = (h16)d[v];
  }
}

// ---------------------------------------------------------------------------
// Kernel 0: pre-transpose RPE tables to f16 [h][176][32] (zero-padded in t)
// One-shot tiny kernel so k_attn can async-copy tables straight to LDS.
// ---------------------------------------------------------------------------
__global__ __launch_bounds__(256) void k_tab(const float* __restrict__ qtab,
                                             const float* __restrict__ ktab,
                                             h16* __restrict__ qtT,
                                             h16* __restrict__ ktT) {
  const int h = blockIdx.x;
  const float* qtp = qtab + (size_t)h * HDIM * T_RPE;
  const float* ktp = ktab + (size_t)h * HDIM * T_RPE;
  h16* qo = qtT + (size_t)h * 176 * HDIM;
  h16* ko = ktT + (size_t)h * 176 * HDIM;
  for (int i = threadIdx.x; i < 176 * HDIM; i += 256) {
    int t = i >> 5, d = i & 31;
    qo[t * HDIM + d] = (t < T_RPE) ? (h16)qtp[d * T_RPE + t] : (h16)0.f;
    ko[t * HDIM + d] = (t < T_RPE) ? (h16)ktp[d * T_RPE + t] : (h16)0.f;
  }
}

// ---------------------------------------------------------------------------
// Kernel 1: qkv = x @ qkv_w + qkv_b  (200704x192 @ 192x576)
// writes q (scaled, f16, [b,h,n,d]), k (f16, [b,h,n,d]), v^T (f16, [b,h,d,n] stride 56)
// ---------------------------------------------------------------------------
__global__ __launch_bounds__(256) void k_qkv(const float* __restrict__ x,
                                             const float* __restrict__ w,
                                             const float* __restrict__ bvec,
                                             h16* __restrict__ qh,
                                             h16* __restrict__ kh,
                                             h16* __restrict__ vt) {
  __shared__ h16 xs[64 * 40];
  __shared__ h16 ws[64 * 40];
  const int m0  = blockIdx.y * 64;
  const int n0  = blockIdx.x * 64;
  const int tid = threadIdx.x, wv = tid >> 5, lane = tid & 31;
  v8f acc[2] = {};
  for (int k0 = 0; k0 < CDIM; k0 += 32) {
    __syncthreads();
    // x tile: 64 rows x 32 cols, vectorized float4 -> half4
    for (int i = tid; i < 64 * 8; i += 256) {
      int r = i >> 3, c4 = (i & 7) << 2;
      v4f f = *(const v4f*)(x + (size_t)(m0 + r) * CDIM + k0 + c4);
      *(v4h*)(xs + r * 40 + c4) = __builtin_convertvector(f, v4h);
    }
    // w tile as B^T[n][k]: coalesced float4 along n
    for (int i = tid; i < 32 * 16; i += 256) {
      int c = i >> 4, n4 = (i & 15) << 2;
      v4f f = *(const v4f*)(w + (size_t)(k0 + c) * (3 * CDIM) + n0 + n4);
      v4h hv = __builtin_convertvector(f, v4h);
      ws[(n4 + 0) * 40 + c] = hv[0];
      ws[(n4 + 1) * 40 + c] = hv[1];
      ws[(n4 + 2) * 40 + c] = hv[2];
      ws[(n4 + 3) * 40 + c] = hv[3];
    }
    if (k0 + 32 < CDIM && tid < 64)  // prefetch next K-slice of x (per row)
      __builtin_prefetch(x + (size_t)(m0 + tid) * CDIM + k0 + 32, 0, 1);
    __syncthreads();
    #pragma unroll
    for (int s = 0; s < 2; ++s) {
      int f = wv * 2 + s;
      acc[s] = wmma_f16(frag_a(xs, 40, (f & 3) * 16, 0, lane),
                        frag_b(ws, 40, (f >> 2) * 16, 0, lane), acc[s]);
    }
  }
  #pragma unroll
  for (int s = 0; s < 2; ++s) {
    int f  = wv * 2 + s;
    int fm = (f & 3) * 16, fn = (f >> 2) * 16;
    int c  = lane & 15, mb = (lane >> 4) << 3;
    int gcol = n0 + fn + c;
    int sec  = gcol / CDIM;                  // 0=q, 1=k, 2=v
    int cc   = gcol % CDIM;
    int hh   = cc >> 5, dd = cc & 31;
    float bv = bvec[gcol];
    #pragma unroll
    for (int v = 0; v < 8; ++v) {
      int g = m0 + fm + mb + v;
      int b = g / N_TOK, n = g % N_TOK;
      float val = acc[s][v] + bv;
      size_t bh = (size_t)(b * NHEAD + hh);
      if (sec == 0)      qh[(bh * N_TOK + n) * HDIM + dd] = (h16)(val * QK_SCALE);
      else if (sec == 1) kh[(bh * N_TOK + n) * HDIM + dd] = (h16)val;
      else               vt[(bh * HDIM + dd) * VSTR + n] = (h16)val;
    }
  }
}

// ---------------------------------------------------------------------------
// Kernel 2: per (window, head) attention with contextual RPE bias + softmax.
// All f16 operand staging uses async global->LDS (ASYNCcnt).
// LDS time-multiplexed: {qs,ks,qts,kts} phase1-2 reused as {Ps,vts} phase3-4.
// ---------------------------------------------------------------------------
__global__ __launch_bounds__(128) void k_attn(const h16* __restrict__ qh,
                                              const h16* __restrict__ kh,
                                              const h16* __restrict__ vt,
                                              const h16* __restrict__ qtT,
                                              const h16* __restrict__ ktT,
                                              const float* __restrict__ mask,
                                              float* __restrict__ attn,
                                              h16* __restrict__ y) {
  __shared__ h16 sm[31352];                  // 62704 B
  h16* qs  = sm;                             // 64 x 40   (byte off 0)
  h16* ks  = sm + 2560;                      // 64 x 40   (byte off 5120)
  h16* qts = sm + 5120;                      // 176 x 40  (byte off 10240)
  h16* kts = sm + 12160;                     // 176 x 40  (byte off 24320)
  h16* Ps  = sm;                             // 64 x 72   (reuse)
  h16* vts = sm + 4608;                      // 32 x 72   (byte off 9216, reuse)
  h16* bsm = sm + 19200;                     // 49 x 176
  h16* Ss  = sm + 27824;                     // 49 x 72

  const int bid = blockIdx.x;
  const int b = bid / NHEAD, h = bid % NHEAD;
  const int tid = threadIdx.x, wv = tid >> 5, lane = tid & 31;
  const unsigned smb = (unsigned)(size_t)sm;

  const h16* qp  = qh  + (size_t)bid * (N_TOK * HDIM);
  const h16* kp  = kh  + (size_t)bid * (N_TOK * HDIM);
  const h16* qtp = qtT + (size_t)h * (176 * HDIM);
  const h16* ktp = ktT + (size_t)h * (176 * HDIM);

  // ---- phase 1: async-stage q,k (49x32) and transposed tables (176x32) ----
  // chunks of 16B: q 196, k 196, qts 704, kts 704  -> 1800 total
  for (int i = tid; i < 1800; i += 128) {
    const h16* src; unsigned dst;
    if (i < 392) {
      int w2 = (i >= 196) ? 1 : 0, ci = i - 196 * w2;
      int r = ci >> 2, c = ci & 3;
      src = (w2 ? kp : qp) + r * HDIM + c * 8;
      dst = smb + (w2 ? 5120u : 0u) + (unsigned)(r * 80 + c * 16);
    } else {
      int j = i - 392;
      int w2 = (j >= 704) ? 1 : 0, ci = j - 704 * w2;
      int t = ci >> 2, c = ci & 3;
      src = (w2 ? ktp : qtp) + t * HDIM + c * 8;
      dst = smb + (w2 ? 24320u : 10240u) + (unsigned)(t * 80 + c * 16);
    }
    async_b128(dst, src);
  }
  for (int i = tid; i < 15 * 32; i += 128) {      // zero-pad q,k rows 49..63
    int r = 49 + (i >> 5), c = i & 31;
    qs[r * 40 + c] = (h16)0.f;
    ks[r * 40 + c] = (h16)0.f;
  }
  wait_async0();
  __syncthreads();

  // ---- phase 2: bsum = q@qtab + k@ktab (fused acc), S = q@k^T ----
  for (int t = wv; t < 44; t += 4) {
    int mt = (t & 3) * 16, nt = (t >> 2) * 16;
    v8f c = {};
    c = wmma_f16(frag_a(qs, 40, mt, 0, lane), frag_b(qts, 40, nt, 0, lane), c);
    c = wmma_f16(frag_a(ks, 40, mt, 0, lane), frag_b(kts, 40, nt, 0, lane), c);
    store_d_h(bsm, 176, mt, nt, c, lane, N_TOK);
  }
  for (int t = wv; t < 16; t += 4) {
    int mt = (t & 3) * 16, nt = (t >> 2) * 16;
    v8f c = {};
    c = wmma_f16(frag_a(qs, 40, mt, 0, lane), frag_b(ks, 40, nt, 0, lane), c);
    store_d_h(Ss, 72, mt, nt, c, lane, N_TOK);
  }
  __syncthreads();

  // ---- phase 3: async-stage v^T, zero-pad P, softmax with RPE gather ----
  const h16* vp = vt + (size_t)bid * (HDIM * VSTR);
  for (int i = tid; i < 224; i += 128) {          // 32 rows x 7 chunks (56 halves)
    int d = i / 7, c = i - d * 7;
    async_b128(smb + 9216u + (unsigned)(d * 144 + c * 16), vp + d * VSTR + c * 8);
  }
  for (int i = tid; i < 32 * 8; i += 128) {       // zero v^T cols 56..63
    int d = i >> 3, c = 56 + (i & 7);
    vts[d * 72 + c] = (h16)0.f;
  }
  for (int i = tid; i < 64 * 64; i += 128) {      // zero-pad P outside 49x49
    int r = i >> 6, c = i & 63;
    if (r >= N_TOK || c >= N_TOK) Ps[r * 72 + c] = (h16)0.f;
  }
  if (tid < N_TOK) {
    const int i  = tid;
    const int ir = i / 7, ic = i % 7;
    const float* mrow = mask + ((size_t)(b & (NWIN - 1)) * N_TOK + i) * N_TOK;
    float* arow = attn + ((size_t)bid * N_TOK + i) * N_TOK;
    float e[N_TOK];
    float mx = -1e30f;
    #pragma unroll
    for (int j = 0; j < N_TOK; ++j) {
      int jr = j / 7, jc = j % 7;
      int t = (ir - jr + 6) * 13 + (ic - jc + 6);   // Swin relative index
      float s = (float)Ss[i * 72 + j] + (float)bsm[i * 176 + t] + mrow[j];
      e[j] = s;
      mx = fmaxf(mx, s);
    }
    float sum = 0.f;
    #pragma unroll
    for (int j = 0; j < N_TOK; ++j) { e[j] = __expf(e[j] - mx); sum += e[j]; }
    float inv = 1.f / sum;
    #pragma unroll
    for (int j = 0; j < N_TOK; ++j) {
      float p = e[j] * inv;
      arow[j] = p;                                  // f32 attn output
      Ps[i * 72 + j] = (h16)p;                      // f16 for P@V
    }
  }
  wait_async0();
  __syncthreads();

  // ---- phase 4: y = P @ V ----
  for (int t = wv; t < 8; t += 4) {
    int mt = (t & 3) * 16, nt = (t >> 2) * 16;
    v8f c = {};
    c = wmma_f16(frag_a(Ps, 72, mt, 0,  lane), frag_b(vts, 72, nt, 0,  lane), c);
    c = wmma_f16(frag_a(Ps, 72, mt, 32, lane), frag_b(vts, 72, nt, 32, lane), c);
    int cc = lane & 15, mb = (lane >> 4) << 3;
    int d  = nt + cc;
    #pragma unroll
    for (int v = 0; v < 8; ++v) {
      int i = mt + mb + v;
      if (i < N_TOK)
        y[(size_t)(b * N_TOK + i) * CDIM + h * HDIM + d] = (h16)c[v];
    }
  }
}

// ---------------------------------------------------------------------------
// Kernel 3: out = y @ proj_w + proj_b  (200704x192 @ 192x192, f32 out)
// f16 A-tile staged with async global->LDS; weights converted via VALU.
// ---------------------------------------------------------------------------
__global__ __launch_bounds__(256) void k_proj(const h16* __restrict__ y,
                                              const float* __restrict__ w,
                                              const float* __restrict__ bvec,
                                              float* __restrict__ out) {
  __shared__ h16 xs[64 * 40];
  __shared__ h16 ws[64 * 40];
  const int m0  = blockIdx.y * 64;
  const int n0  = blockIdx.x * 64;
  const int tid = threadIdx.x, wv = tid >> 5, lane = tid & 31;
  const unsigned xsb = (unsigned)(size_t)xs;
  v8f acc[2] = {};
  for (int k0 = 0; k0 < CDIM; k0 += 32) {
    __syncthreads();
    // y tile (f16): 64 rows x 4 x 16B chunks, pure async copy
    {
      int r = tid >> 2, c = tid & 3;                 // 256 threads = 256 chunks
      async_b128(xsb + (unsigned)(r * 80 + c * 16),
                 y + (size_t)(m0 + r) * CDIM + k0 + c * 8);
    }
    for (int i = tid; i < 32 * 16; i += 256) {       // proj_w as B^T, float4 along n
      int c = i >> 4, n4 = (i & 15) << 2;
      v4f f = *(const v4f*)(w + (size_t)(k0 + c) * CDIM + n0 + n4);
      v4h hv = __builtin_convertvector(f, v4h);
      ws[(n4 + 0) * 40 + c] = hv[0];
      ws[(n4 + 1) * 40 + c] = hv[1];
      ws[(n4 + 2) * 40 + c] = hv[2];
      ws[(n4 + 3) * 40 + c] = hv[3];
    }
    wait_async0();
    __syncthreads();
    #pragma unroll
    for (int s = 0; s < 2; ++s) {
      int f = wv * 2 + s;
      acc[s] = wmma_f16(frag_a(xs, 40, (f & 3) * 16, 0, lane),
                        frag_b(ws, 40, (f >> 2) * 16, 0, lane), acc[s]);
    }
  }
  #pragma unroll
  for (int s = 0; s < 2; ++s) {
    int f  = wv * 2 + s;
    int fm = (f & 3) * 16, fn = (f >> 2) * 16;
    int c  = lane & 15, mb = (lane >> 4) << 3;
    int gcol = n0 + fn + c;
    float bv = bvec[gcol];
    #pragma unroll
    for (int v = 0; v < 8; ++v)
      out[(size_t)(m0 + fm + mb + v) * CDIM + gcol] = acc[s][v] + bv;
  }
}

// ---------------------------------------------------------------------------
extern "C" void kernel_launch(void* const* d_in, const int* in_sizes, int n_in,
                              void* d_out, int out_size, void* d_ws, size_t ws_size,
                              hipStream_t stream) {
  (void)in_sizes; (void)n_in; (void)out_size; (void)ws_size;
  const float* x      = (const float*)d_in[0];
  const float* mask   = (const float*)d_in[1];
  const float* qkv_w  = (const float*)d_in[2];
  const float* qkv_b  = (const float*)d_in[3];
  const float* qtab   = (const float*)d_in[4];
  const float* ktab   = (const float*)d_in[5];
  const float* proj_w = (const float*)d_in[6];
  const float* proj_b = (const float*)d_in[7];
  // d_in[8] (rpe_indices) is recomputed arithmetically in-kernel.

  float* out  = (float*)d_out;
  float* attn = out + (size_t)NROW * CDIM;           // outputs concatenated

  const size_t QSZ = (size_t)BWIN * NHEAD * N_TOK * HDIM;   // 38,535,168 halves
  const size_t VSZ = (size_t)BWIN * NHEAD * HDIM * VSTR;    // 44,040,192 halves
  const size_t TSZ = (size_t)NHEAD * 176 * HDIM;            // 33,792 halves
  h16* qh  = (h16*)d_ws;
  h16* kh  = qh + QSZ;
  h16* vt  = kh + QSZ;
  h16* yb  = vt + VSZ;
  h16* qtT = yb + QSZ;
  h16* ktT = qtT + TSZ;

  k_tab <<<dim3(NHEAD),        256, 0, stream>>>(qtab, ktab, qtT, ktT);
  k_qkv <<<dim3(9, NROW / 64), 256, 0, stream>>>(x, qkv_w, qkv_b, qh, kh, vt);
  k_attn<<<dim3(BWIN * NHEAD), 128, 0, stream>>>(qh, kh, vt, qtT, ktT, mask, attn, yb);
  k_proj<<<dim3(3, NROW / 64), 256, 0, stream>>>(yb, proj_w, proj_b, out);
}
